// BasewiseTransformerEmbedder_30829275250975
// MI455X (gfx1250) — matrix operand order repaired and checked by reading, not compiled, not run on hardware
//
#include <hip/hip_runtime.h>
#include <hip/hip_bf16.h>

typedef __attribute__((ext_vector_type(16))) _Float16 v16h;
typedef __attribute__((ext_vector_type(8)))  _Float16 v8h;
typedef __attribute__((ext_vector_type(8)))  float    v8f;

#define B_   4
#define T_   256
#define L_   128
#define D_   64
#define H_   4
#define NL_  2
#define HID_ 128
#define HD_  16
#define LN_EPS 1e-5f

#define NWAVES 8
#define NTHR   (NWAVES * 32)

// ---------------- WMMA operand loaders (layouts per CDNA5 ISA) ----

// A operand, 16x32 f16, from row-major [M x ld] f16. lane = M row (lo half-wave
// holds K 0..7 / 16..23 runs, hi half-wave K 8..15 / 24..31).
static __device__ inline v16h ldA(const _Float16* base, int ld, int m0, int k0, int lane) {
  int m  = m0 + (lane & 15);
  int hf = lane >> 4;
  union { v16h v; v8h h[2]; } u;
  const _Float16* p = base + m * ld + k0 + hf * 8;
  u.h[0] = *(const v8h*)(p);
  u.h[1] = *(const v8h*)(p + 16);
  return u.v;
}

// A operand with only K=0..15 valid (HD=16 padded to 32 with zeros).
static __device__ inline v16h ldA_k16(const _Float16* base, int ld, int m0, int k0, int lane) {
  int m  = m0 + (lane & 15);
  int hf = lane >> 4;
  union { v16h v; v8h h[2]; } u;
  u.h[0] = *(const v8h*)(base + m * ld + k0 + hf * 8);
#pragma unroll
  for (int i = 0; i < 8; ++i) u.h[1][i] = (_Float16)0.0f;
  return u.v;
}

// B operand, 32x16 f16, from row-major [K x ld] f16. lane%16 = N column; K runs
// like A (lo half-wave K 0..7 / 16..23, hi half-wave K 8..15 / 24..31).
static __device__ inline v16h ldB(const _Float16* base, int ld, int k0, int n0, int lane) {
  v16h v;
  const _Float16* p = base + (k0 + (lane >> 4) * 8) * ld + n0 + (lane & 15);
#pragma unroll
  for (int i = 0; i < 8; ++i) { v[i] = p[i * ld]; v[8 + i] = p[(16 + i) * ld]; }
  return v;
}

// B operand with only K=0..15 valid (rows 16..31 zero).
static __device__ inline v16h ldB_k16(const _Float16* base, int ld, int n0, int lane) {
  v16h v;
  const _Float16* p = base + ((lane >> 4) * 8) * ld + n0 + (lane & 15);
#pragma unroll
  for (int i = 0; i < 8; ++i) { v[i] = p[i * ld]; v[8 + i] = (_Float16)0.0f; }
  return v;
}

static __device__ inline v8f wmma16(v16h a, v16h b, v8f c) {
  return __builtin_amdgcn_wmma_f32_16x16x32_f16(false, a, false, b, (short)0, c, false, false);
}

// ---------------- weight f32 -> f16 conversion (one-time, into d_ws) --------

__global__ void bte_convert_weights(const float* __restrict__ qkv_w,
                                    const float* __restrict__ out_w,
                                    const float* __restrict__ mlp_w,
                                    _Float16* __restrict__ qkvw16,
                                    _Float16* __restrict__ outw16,
                                    _Float16* __restrict__ mlpw16) {
  const int nq = NL_ * D_ * 3 * D_;   // 24576
  const int no = NL_ * D_ * D_;       // 8192
  int i = blockIdx.x * blockDim.x + threadIdx.x;
  if (i < nq) qkvw16[i] = (_Float16)qkv_w[i];
  if (i < no) {
    outw16[i] = (_Float16)out_w[i];
    mlpw16[i] = (_Float16)mlp_w[i];
  }
}

// ---------------- main kernel: one workgroup == one sequence ----------------

__global__ __launch_bounds__(NTHR) void bte_kernel(
    const int*   __restrict__ base_ids,
    const float* __restrict__ emb,
    const float* __restrict__ qkv_b,
    const float* __restrict__ out_b,
    const float* __restrict__ ln1_g, const float* __restrict__ ln1_b,
    const float* __restrict__ mlp_b,
    const float* __restrict__ ln2_g, const float* __restrict__ ln2_b,
    const float* __restrict__ proj_w, const float* __restrict__ proj_b,
    const _Float16* __restrict__ qkvw16,
    const _Float16* __restrict__ outw16,
    const _Float16* __restrict__ mlpw16,
    float* __restrict__ out) {
  extern __shared__ char smem[];
  _Float16* x16   = (_Float16*)(smem);            // 128x64   f16 activations
  _Float16* qkv16 = (_Float16*)(smem + 16384);    // 128x192  f16 qkv
  _Float16* kT16  = (_Float16*)(smem + 65536);    // [4][16][128] K transposed
  float*    scf   = (float*)   (smem + 81920);    // 128x128  f32 scores / 128x64 f32 pre-LN
  _Float16* p16   = (_Float16*)(smem + 147456);   // 128x128  f16 probs
  _Float16* a16   = (_Float16*)(smem + 180224);   // 128x64   f16 attn out
  _Float16* g16   = (_Float16*)(smem + 196608);   // 128x64   f16 LN1 out

  const int n    = blockIdx.x;            // sequence index (0..1023)
  const int tid  = threadIdx.x;
  const int wave = tid >> 5;
  const int lane = tid & 31;
  const int nlo  = lane & 15;
  const int mhi  = (lane >> 4) << 3;

  // ---- embedding gather: x = emb[base_ids[n]] ----
  for (int idx = tid; idx < L_ * D_; idx += NTHR) {
    int t = idx >> 6, d = idx & 63;
    int tok = base_ids[n * L_ + t];
    x16[idx] = (_Float16)emb[tok * D_ + d];
  }
  __syncthreads();

  for (int layer = 0; layer < NL_; ++layer) {
    // ---- GEMM1: qkv = x(128x64) @ qkv_w(64x192) + qkv_b ----
    const _Float16* wq = qkvw16 + layer * D_ * 3 * D_;
    for (int tile = wave; tile < 8 * 12; tile += NWAVES) {
      int mi = tile / 12, ni = tile % 12;
      v8f c = {};
#pragma unroll
      for (int kk = 0; kk < 2; ++kk) {
        v16h a = ldA(x16, D_, mi * 16, kk * 32, lane);
        v16h b = ldB(wq, 3 * D_, kk * 32, ni * 16, lane);
        c = wmma16(a, b, c);
      }
      float bv = qkv_b[layer * 3 * D_ + ni * 16 + nlo];
      int nn = ni * 16 + nlo, mb = mi * 16 + mhi;
#pragma unroll
      for (int r = 0; r < 8; ++r)
        qkv16[(mb + r) * (3 * D_) + nn] = (_Float16)(c[r] + bv);
    }
    __syncthreads();

    // ---- build K^T per head: kT[h][d][t] ----
    for (int idx = tid; idx < H_ * HD_ * L_; idx += NTHR) {
      int t = idx & (L_ - 1);
      int d = (idx >> 7) & (HD_ - 1);
      int h = idx >> 11;
      kT16[idx] = qkv16[t * (3 * D_) + D_ + h * HD_ + d];
    }
    __syncthreads();

    // ---- attention, one head at a time ----
    for (int h = 0; h < H_; ++h) {
      // scores = Q_h(128x16) @ K_h^T(16x128), K padded to 32
      for (int tile = wave; tile < 64; tile += NWAVES) {
        int mi = tile >> 3, ni = tile & 7;
        v16h a = ldA_k16(qkv16, 3 * D_, mi * 16, h * HD_, lane);
        v16h b = ldB_k16(kT16 + h * HD_ * L_, L_, ni * 16, lane);
        v8f c = {};
        c = wmma16(a, b, c);
        int nn = ni * 16 + nlo, mb = mi * 16 + mhi;
#pragma unroll
        for (int r = 0; r < 8; ++r) scf[(mb + r) * L_ + nn] = c[r];
      }
      __syncthreads();

      // softmax(scores*0.25 + alibi) per row; probs -> f16
      if (tid < L_) {
        const float slope = exp2f(-2.0f * (float)h);   // 0.25^h
        float mx = -1e30f;
        for (int k = 0; k < L_; ++k) {
          float v = scf[tid * L_ + k] * 0.25f - slope * fabsf((float)(k - tid));
          mx = fmaxf(mx, v);
        }
        float s = 0.0f;
        for (int k = 0; k < L_; ++k) {
          float v = scf[tid * L_ + k] * 0.25f - slope * fabsf((float)(k - tid));
          float e = __expf(v - mx);
          scf[tid * L_ + k] = e;
          s += e;
        }
        float inv = 1.0f / s;
        for (int k = 0; k < L_; ++k)
          p16[tid * L_ + k] = (_Float16)(scf[tid * L_ + k] * inv);
      }
      __syncthreads();

      // attn_h = probs(128x128) @ V_h(128x16)
      for (int tile = wave; tile < 8; tile += NWAVES) {
        int mi = tile;
        v8f c = {};
#pragma unroll
        for (int kk = 0; kk < 4; ++kk) {
          v16h a = ldA(p16, L_, mi * 16, kk * 32, lane);
          v16h b = ldB(qkv16, 3 * D_, kk * 32, 2 * D_ + h * HD_, lane);
          c = wmma16(a, b, c);
        }
        int nn = h * HD_ + nlo, mb = mi * 16 + mhi;
#pragma unroll
        for (int r = 0; r < 8; ++r) a16[(mb + r) * D_ + nn] = (_Float16)c[r];
      }
      __syncthreads();
    }

    // ---- GEMM2: attn(128x64) @ out_w(64x64) + out_b -> f32 (reuse scf) ----
    float* hf = scf;
    const _Float16* wo = outw16 + layer * D_ * D_;
    for (int tile = wave; tile < 8 * 4; tile += NWAVES) {
      int mi = tile >> 2, ni = tile & 3;
      v8f c = {};
#pragma unroll
      for (int kk = 0; kk < 2; ++kk) {
        v16h a = ldA(a16, D_, mi * 16, kk * 32, lane);
        v16h b = ldB(wo, D_, kk * 32, ni * 16, lane);
        c = wmma16(a, b, c);
      }
      float bv = out_b[layer * D_ + ni * 16 + nlo];
      int nn = ni * 16 + nlo, mb = mi * 16 + mhi;
#pragma unroll
      for (int r = 0; r < 8; ++r) hf[(mb + r) * D_ + nn] = c[r] + bv;
    }
    __syncthreads();

    // ---- LN1 -> g16 ----
    if (tid < L_) {
      float m = 0.0f;
      for (int d = 0; d < D_; ++d) m += hf[tid * D_ + d];
      m *= (1.0f / D_);
      float v = 0.0f;
      for (int d = 0; d < D_; ++d) { float u = hf[tid * D_ + d] - m; v += u * u; }
      float rs = rsqrtf(v * (1.0f / D_) + LN_EPS);
      for (int d = 0; d < D_; ++d)
        g16[tid * D_ + d] = (_Float16)((hf[tid * D_ + d] - m) * rs * ln1_g[layer * D_ + d] +
                                       ln1_b[layer * D_ + d]);
    }
    __syncthreads();

    // ---- GEMM3: relu(h @ mlp_w + mlp_b) -> f32 hf ----
    const _Float16* wm = mlpw16 + layer * D_ * D_;
    for (int tile = wave; tile < 8 * 4; tile += NWAVES) {
      int mi = tile >> 2, ni = tile & 3;
      v8f c = {};
#pragma unroll
      for (int kk = 0; kk < 2; ++kk) {
        v16h a = ldA(g16, D_, mi * 16, kk * 32, lane);
        v16h b = ldB(wm, D_, kk * 32, ni * 16, lane);
        c = wmma16(a, b, c);
      }
      float bv = mlp_b[layer * D_ + ni * 16 + nlo];
      int nn = ni * 16 + nlo, mb = mi * 16 + mhi;
#pragma unroll
      for (int r = 0; r < 8; ++r) {
        float val = c[r] + bv;
        hf[(mb + r) * D_ + nn] = val > 0.0f ? val : 0.0f;
      }
    }
    __syncthreads();

    // ---- LN2 -> x16 (next layer input) ----
    if (tid < L_) {
      float m = 0.0f;
      for (int d = 0; d < D_; ++d) m += hf[tid * D_ + d];
      m *= (1.0f / D_);
      float v = 0.0f;
      for (int d = 0; d < D_; ++d) { float u = hf[tid * D_ + d] - m; v += u * u; }
      float rs = rsqrtf(v * (1.0f / D_) + LN_EPS);
      for (int d = 0; d < D_; ++d)
        x16[tid * D_ + d] = (_Float16)((hf[tid * D_ + d] - m) * rs * ln2_g[layer * D_ + d] +
                                       ln2_b[layer * D_ + d]);
    }
    __syncthreads();
  }

  // ---- final projection: out[n] = x[:, -1, :] @ proj_w + proj_b ----
  if (tid < HID_) {
    float s = proj_b[tid];
    for (int d = 0; d < D_; ++d)
      s += (float)x16[(L_ - 1) * D_ + d] * proj_w[d * HID_ + tid];
    out[n * HID_ + tid] = s;
  }
}

// ---------------------------------------------------------------------------

extern "C" void kernel_launch(void* const* d_in, const int* in_sizes, int n_in,
                              void* d_out, int out_size, void* d_ws, size_t ws_size,
                              hipStream_t stream) {
  (void)in_sizes; (void)n_in; (void)out_size; (void)ws_size;
  const int*   base_ids = (const int*)  d_in[0];
  const float* emb      = (const float*)d_in[1];
  const float* qkv_w    = (const float*)d_in[2];
  const float* qkv_b    = (const float*)d_in[3];
  const float* out_w    = (const float*)d_in[4];
  const float* out_b    = (const float*)d_in[5];
  const float* ln1_g    = (const float*)d_in[6];
  const float* ln1_b    = (const float*)d_in[7];
  const float* mlp_w    = (const float*)d_in[8];
  const float* mlp_b    = (const float*)d_in[9];
  const float* ln2_g    = (const float*)d_in[10];
  const float* ln2_b    = (const float*)d_in[11];
  const float* proj_w   = (const float*)d_in[12];
  const float* proj_b   = (const float*)d_in[13];
  float* out = (float*)d_out;

  _Float16* qkvw16 = (_Float16*)d_ws;                         // 24576 halves
  _Float16* outw16 = (_Float16*)((char*)d_ws + 49152);        //  8192 halves
  _Float16* mlpw16 = (_Float16*)((char*)d_ws + 65536);        //  8192 halves

  bte_convert_weights<<<(NL_ * D_ * 3 * D_ + 255) / 256, 256, 0, stream>>>(
      qkv_w, out_w, mlp_w, qkvw16, outw16, mlpw16);

  const size_t shmem = 212992;  // 208 KB dynamic LDS (<= 320 KB/WGP)
  (void)hipFuncSetAttribute((const void*)bte_kernel,
                            hipFuncAttributeMaxDynamicSharedMemorySize,
                            (int)shmem);
  bte_kernel<<<B_ * T_, NTHR, shmem, stream>>>(
      base_ids, emb, qkv_b, out_b, ln1_g, ln1_b, mlp_b, ln2_g, ln2_b,
      proj_w, proj_b, qkvw16, outw16, mlpw16, out);
}
